// LayerNormLSTMCell_66718021976166
// MI455X (gfx1250) — matrix operand order repaired and checked by reading, not compiled, not run on hardware
//
#include <hip/hip_runtime.h>
#include <math.h>

// LayerNorm LSTM cell for MI455X (gfx1250, wave32, WMMA).
// ws requirement: 2 * 8192 * 4096 * 4 bytes = 256 MiB (P and Q projection buffers).

#define B_ROWS 8192
#define NCOLS  4096
#define HDIM   1024
#define LN_EPS 1e-5f

#define BM 128
#define BN 128
#define BK 32
#define LSTRIDE 28   // padded uint stride per LDS row (112B: 16B-aligned, conflict-free)

typedef __attribute__((ext_vector_type(16))) __bf16 v16bf;
typedef __attribute__((ext_vector_type(8)))  float  v8f;

union Frag16 { v16bf v; uint4 q[2]; };

// Pack two floats' truncated-bf16 halves into one dword (elem0 in low 16 bits).
__device__ __forceinline__ unsigned pack_bf16_trunc(float a, float b) {
  return (__float_as_uint(a) >> 16) | (__float_as_uint(b) & 0xffff0000u);
}
__device__ __forceinline__ float bf16_hi_val(float a) {
  return __uint_as_float(__float_as_uint(a) & 0xffff0000u);
}
// hi = trunc16(x,y) packed; lo = trunc16(x - hi(x), y - hi(y)) packed.
__device__ __forceinline__ void cvt_pair(float x, float y, unsigned& hi, unsigned& lo) {
  hi = pack_bf16_trunc(x, y);
  lo = pack_bf16_trunc(x - bf16_hi_val(x), y - bf16_hi_val(y));
}

// C[M,N] = A[M,K] * W[N,K]^T  via bf16x3 (Ah*Bh + Ah*Bl + Al*Bh), f32 accumulate.
// Grid: (N/BN, M/BM), block: 256 threads (8 waves: 4 along M x 2 along N).
__global__ __launch_bounds__(256)
void gemm_bf16x3(const float* __restrict__ A, const float* __restrict__ W,
                 float* __restrict__ C, int K, int N) {
  __shared__ unsigned sAhi[BM][LSTRIDE];
  __shared__ unsigned sAlo[BM][LSTRIDE];
  __shared__ unsigned sBhi[BN][LSTRIDE];
  __shared__ unsigned sBlo[BN][LSTRIDE];

  const int t    = threadIdx.x;
  const int lane = t & 31;
  const int wave = t >> 5;        // 0..7
  const int wm   = wave & 3;      // 0..3 (M direction, 32 rows each)
  const int wn   = wave >> 2;     // 0..1 (N direction, 64 cols each)
  const int blockM = blockIdx.y * BM;
  const int blockN = blockIdx.x * BN;

  // Cooperative fill mapping: each thread fills half a row (16 f32) of A and B tiles.
  const int fr = t >> 1;          // 0..127
  const int fh = t & 1;           // 0/1 -> k-offset 0/16
  const float* aSrc = A + (size_t)(blockM + fr) * K + fh * 16;
  const float* bSrc = W + (size_t)(blockN + fr) * K + fh * 16;

  v8f acc[2][4];
#pragma unroll
  for (int sm = 0; sm < 2; ++sm)
#pragma unroll
    for (int sn = 0; sn < 4; ++sn)
      acc[sm][sn] = (v8f){0.f,0.f,0.f,0.f,0.f,0.f,0.f,0.f};

  for (int k0 = 0; k0 < K; k0 += BK) {
    // ---- stage: global f32 -> bf16 hi/lo in LDS ----
    const float4* ap = (const float4*)(aSrc + k0);
    const float4* bp = (const float4*)(bSrc + k0);
    float4 av0 = ap[0], av1 = ap[1], av2 = ap[2], av3 = ap[3];
    float4 bv0 = bp[0], bv1 = bp[1], bv2 = bp[2], bv3 = bp[3];

    unsigned ahi[8] __attribute__((aligned(16)));
    unsigned alo[8] __attribute__((aligned(16)));
    unsigned bhi[8] __attribute__((aligned(16)));
    unsigned blo[8] __attribute__((aligned(16)));
    cvt_pair(av0.x, av0.y, ahi[0], alo[0]);  cvt_pair(av0.z, av0.w, ahi[1], alo[1]);
    cvt_pair(av1.x, av1.y, ahi[2], alo[2]);  cvt_pair(av1.z, av1.w, ahi[3], alo[3]);
    cvt_pair(av2.x, av2.y, ahi[4], alo[4]);  cvt_pair(av2.z, av2.w, ahi[5], alo[5]);
    cvt_pair(av3.x, av3.y, ahi[6], alo[6]);  cvt_pair(av3.z, av3.w, ahi[7], alo[7]);
    cvt_pair(bv0.x, bv0.y, bhi[0], blo[0]);  cvt_pair(bv0.z, bv0.w, bhi[1], blo[1]);
    cvt_pair(bv1.x, bv1.y, bhi[2], blo[2]);  cvt_pair(bv1.z, bv1.w, bhi[3], blo[3]);
    cvt_pair(bv2.x, bv2.y, bhi[4], blo[4]);  cvt_pair(bv2.z, bv2.w, bhi[5], blo[5]);
    cvt_pair(bv3.x, bv3.y, bhi[6], blo[6]);  cvt_pair(bv3.z, bv3.w, bhi[7], blo[7]);

    *(uint4*)&sAhi[fr][fh*8]     = *(const uint4*)&ahi[0];
    *(uint4*)&sAhi[fr][fh*8 + 4] = *(const uint4*)&ahi[4];
    *(uint4*)&sAlo[fr][fh*8]     = *(const uint4*)&alo[0];
    *(uint4*)&sAlo[fr][fh*8 + 4] = *(const uint4*)&alo[4];
    *(uint4*)&sBhi[fr][fh*8]     = *(const uint4*)&bhi[0];
    *(uint4*)&sBhi[fr][fh*8 + 4] = *(const uint4*)&bhi[4];
    *(uint4*)&sBlo[fr][fh*8]     = *(const uint4*)&blo[0];
    *(uint4*)&sBlo[fr][fh*8 + 4] = *(const uint4*)&blo[4];
    __syncthreads();

    // ---- A fragments (ISA 16-bit A 16x32 layout: lanes<16 K=[0,8)+[16,24), lanes>=16 K=[8,16)+[24,32)) ----
    Frag16 fah[2], fal[2];
    const int uc = (lane >> 4) * 4;
#pragma unroll
    for (int sm = 0; sm < 2; ++sm) {
      const int row = wm * 32 + sm * 16 + (lane & 15);
      fah[sm].q[0] = *(const uint4*)&sAhi[row][uc];
      fah[sm].q[1] = *(const uint4*)&sAhi[row][uc + 8];
      fal[sm].q[0] = *(const uint4*)&sAlo[row][uc];
      fal[sm].q[1] = *(const uint4*)&sAlo[row][uc + 8];
    }

    // ---- B fragments (B 32x16, col = lane&15, same K layout as A: K=[8h,8h+8)+[16+8h,24+8h)) + WMMA ----
    const int kb = (lane >> 4) * 4;
#pragma unroll
    for (int sn = 0; sn < 4; ++sn) {
      const int col = wn * 64 + sn * 16 + (lane & 15);
      Frag16 fbh, fbl;
      fbh.q[0] = *(const uint4*)&sBhi[col][kb];
      fbh.q[1] = *(const uint4*)&sBhi[col][kb + 8];
      fbl.q[0] = *(const uint4*)&sBlo[col][kb];
      fbl.q[1] = *(const uint4*)&sBlo[col][kb + 8];
#pragma unroll
      for (int sm = 0; sm < 2; ++sm) {
        acc[sm][sn] = __builtin_amdgcn_wmma_f32_16x16x32_bf16(
            false, fah[sm].v, false, fbh.v, (short)0, acc[sm][sn], false, false);
        acc[sm][sn] = __builtin_amdgcn_wmma_f32_16x16x32_bf16(
            false, fah[sm].v, false, fbl.v, (short)0, acc[sm][sn], false, false);
        acc[sm][sn] = __builtin_amdgcn_wmma_f32_16x16x32_bf16(
            false, fal[sm].v, false, fbh.v, (short)0, acc[sm][sn], false, false);
      }
    }
    __syncthreads();
  }

  // ---- store C (C/D layout: VGPR v -> row v + 8*(lane>>4), col = lane&15) ----
  const int rhalf = (lane >> 4) * 8;
  const int cIdx  = lane & 15;
#pragma unroll
  for (int sm = 0; sm < 2; ++sm)
#pragma unroll
    for (int sn = 0; sn < 4; ++sn) {
      const int gr0 = blockM + wm * 32 + sm * 16 + rhalf;
      const int gc  = blockN + wn * 64 + sn * 16 + cIdx;
#pragma unroll
      for (int v = 0; v < 8; ++v)
        C[(size_t)(gr0 + v) * N + gc] = acc[sm][sn][v];
    }
}

__device__ __forceinline__ float sigmoidf_dev(float x) {
  return 1.0f / (1.0f + __expf(-x));
}

__device__ __forceinline__ float block_reduce_sum(float v, float* s) {
  const int t = threadIdx.x;
  s[t] = v;
  __syncthreads();
#pragma unroll
  for (int off = 128; off > 0; off >>= 1) {
    if (t < off) s[t] += s[t + off];
    __syncthreads();
  }
  float r = s[0];
  __syncthreads();
  return r;
}

// One workgroup per batch row: LN(P)+LN(Q) -> gates -> c_new -> LN(c) -> h_new.
__global__ __launch_bounds__(256)
void lstm_epilogue(const float* __restrict__ P, const float* __restrict__ Q,
                   const float* __restrict__ c_prev,
                   const float* __restrict__ bi, const float* __restrict__ bh,
                   const float* __restrict__ g_in, const float* __restrict__ b_in,
                   const float* __restrict__ g_hid, const float* __restrict__ b_hid,
                   const float* __restrict__ g_cell, const float* __restrict__ b_cell,
                   float* __restrict__ h_out, float* __restrict__ c_out) {
  __shared__ float sp[NCOLS];
  __shared__ float sq[NCOLS];
  __shared__ float sred[256];

  const int t   = threadIdx.x;
  const int row = blockIdx.x;
  const float* Prow = P + (size_t)row * NCOLS;
  const float* Qrow = Q + (size_t)row * NCOLS;

  float ap = 0.f, app = 0.f, aq = 0.f, aqq = 0.f;
#pragma unroll
  for (int u = 0; u < 16; ++u) {
    const int j = u * 256 + t;
    const float p = Prow[j] + bi[j];
    const float q = Qrow[j] + bh[j];
    sp[j] = p; sq[j] = q;
    ap += p; app += p * p; aq += q; aqq += q * q;
  }
  const float sump  = block_reduce_sum(ap,  sred);
  const float sumpp = block_reduce_sum(app, sred);
  const float sumq  = block_reduce_sum(aq,  sred);
  const float sumqq = block_reduce_sum(aqq, sred);
  const float invN = 1.0f / (float)NCOLS;
  const float mup = sump * invN, muq = sumq * invN;
  const float rsp = rsqrtf(fmaxf(sumpp * invN - mup * mup, 0.f) + LN_EPS);
  const float rsq = rsqrtf(fmaxf(sumqq * invN - muq * muq, 0.f) + LN_EPS);

#pragma unroll
  for (int u = 0; u < 16; ++u) {
    const int j = u * 256 + t;
    const float comb = (sp[j] - mup) * rsp * g_in[j] + b_in[j]
                     + (sq[j] - muq) * rsq * g_hid[j] + b_hid[j];
    sp[j] = comb;
  }
  __syncthreads();

  float ac = 0.f, acc2 = 0.f;
  float cnv[4], ogv[4];
#pragma unroll
  for (int u = 0; u < 4; ++u) {
    const int j = u * 256 + t;
    const float ig = sigmoidf_dev(sp[j]);
    const float fg = sigmoidf_dev(sp[HDIM + j]);
    const float gg = tanhf(sp[2 * HDIM + j]);
    const float og = sigmoidf_dev(sp[3 * HDIM + j]);
    const float cn = fg * c_prev[(size_t)row * HDIM + j] + ig * gg;
    c_out[(size_t)row * HDIM + j] = cn;
    cnv[u] = cn; ogv[u] = og;
    ac += cn; acc2 += cn * cn;
  }
  const float sumc  = block_reduce_sum(ac,   sred);
  const float sumcc = block_reduce_sum(acc2, sred);
  const float invH = 1.0f / (float)HDIM;
  const float muc = sumc * invH;
  const float rsc = rsqrtf(fmaxf(sumcc * invH - muc * muc, 0.f) + LN_EPS);

#pragma unroll
  for (int u = 0; u < 4; ++u) {
    const int j = u * 256 + t;
    const float hn = ogv[u] * tanhf((cnv[u] - muc) * rsc * g_cell[j] + b_cell[j]);
    h_out[(size_t)row * HDIM + j] = hn;
  }
}

extern "C" void kernel_launch(void* const* d_in, const int* in_sizes, int n_in,
                              void* d_out, int out_size, void* d_ws, size_t ws_size,
                              hipStream_t stream) {
  const float* x      = (const float*)d_in[0];
  const float* h_prev = (const float*)d_in[1];
  const float* c_prev = (const float*)d_in[2];
  const float* Wi     = (const float*)d_in[3];
  const float* bi     = (const float*)d_in[4];
  const float* Wh     = (const float*)d_in[5];
  const float* bh     = (const float*)d_in[6];
  const float* g_in   = (const float*)d_in[7];
  const float* b_in   = (const float*)d_in[8];
  const float* g_hid  = (const float*)d_in[9];
  const float* b_hid  = (const float*)d_in[10];
  const float* g_cell = (const float*)d_in[11];
  const float* b_cell = (const float*)d_in[12];

  float* P = (float*)d_ws;                       // [8192, 4096] f32
  float* Q = P + (size_t)B_ROWS * NCOLS;         // [8192, 4096] f32
  float* h_out = (float*)d_out;                  // [8192, 1024]
  float* c_out = h_out + (size_t)B_ROWS * HDIM;  // [8192, 1024]

  dim3 blk(256);
  dim3 grid(NCOLS / BN, B_ROWS / BM);
  gemm_bf16x3<<<grid, blk, 0, stream>>>(x,      Wi, P,  512, NCOLS);
  gemm_bf16x3<<<grid, blk, 0, stream>>>(h_prev, Wh, Q, 1024, NCOLS);
  lstm_epilogue<<<B_ROWS, blk, 0, stream>>>(P, Q, c_prev, bi, bh,
                                            g_in, b_in, g_hid, b_hid,
                                            g_cell, b_cell, h_out, c_out);
}
